// LSTMModelTrig_1425929142285
// MI455X (gfx1250) — matrix operand order repaired and checked by reading, not compiled, not run on hardware
//
#include <hip/hip_runtime.h>
#include <stdint.h>
#include <string.h>

typedef _Float16 v16h __attribute__((ext_vector_type(16)));
typedef _Float16 v8h  __attribute__((ext_vector_type(8)));
typedef float    v8f  __attribute__((ext_vector_type(8)));

#define HID    50
#define NFRAG  32     // 16 M-tiles (4 quadrants x 4) * 2 K-chunks
#define WAVES  2
#define HROW   72     // padded halfs per hbuf row

// gfx1250 hardware tanh (trans op)
#if __has_builtin(__builtin_amdgcn_tanhf)
  #define HW_TANH(x) __builtin_amdgcn_tanhf(x)
#elif __has_builtin(__builtin_amdgcn_tanh_f32)
  #define HW_TANH(x) __builtin_amdgcn_tanh_f32(x)
#endif

// in-register 16-lane row swap (D-layout -> B-layout transpose helper)
#if __has_builtin(__builtin_amdgcn_permlane16_swap)
  #define HAVE_PLSWAP 1
#endif

__device__ __forceinline__ float lstm_tanh(float x) {
#ifdef HW_TANH
    return HW_TANH(x);
#else
    float e = __builtin_amdgcn_exp2f(2.8853900817779268f * x);
    return 1.0f - 2.0f * __builtin_amdgcn_rcpf(1.0f + e);
#endif
}
// sigmoid given HALF-scaled preactivation a = x/2 (scaling folded into weights)
__device__ __forceinline__ float lstm_sigmoid_half(float a) {
#ifdef HW_TANH
    return 0.5f + 0.5f * HW_TANH(a);
#else
    float e = __builtin_amdgcn_exp2f(-2.8853900817779268f * a);
    return __builtin_amdgcn_rcpf(1.0f + e);
#endif
}
__device__ __forceinline__ uint32_t pkh(float a, float b) {
    auto r = __builtin_amdgcn_cvt_pkrtz(a, b);   // __fp16 ext_vector(2)
    uint32_t u;
    memcpy(&u, &r, 4);
    return u;
}

// One wave = 16 batch rows; whole recurrence in registers.
// gates^T = A[256x64] @ B[64x16], A = packed {W_hh | W_ih | b_ih+b_hh},
// quadrants (i,f,g,o) padded to 64 rows so all 4 gate values for a
// (batch, hidden) pair land in the same lane / same accumulator slot.
// i,f,o rows pre-scaled by 0.5 for sigmoid-via-tanh.
__global__ __launch_bounds__(WAVES * 32, 1)
void lstm_wmma_kernel(const float* __restrict__ x,
                      const float* __restrict__ W_ih,
                      const float* __restrict__ W_hh,
                      const float* __restrict__ b_ih,
                      const float* __restrict__ b_hh,
                      const float* __restrict__ W_fc,
                      const float* __restrict__ b_fc,
                      float* __restrict__ out,
                      int T)
{
    __shared__ __attribute__((aligned(16))) _Float16 astage[NFRAG][32][16];
    __shared__ __attribute__((aligned(16))) _Float16 hbuf[WAVES][16][HROW];

    const int tid  = threadIdx.x;
    const int w    = tid >> 5;
    const int lane = tid & 31;
    const int n    = lane & 15;   // batch column within tile
    const int hl   = lane >> 4;   // lane half

    // ---- one-time: build packed f16 A fragments in LDS (contiguous k = 16*half + e layout) ----
    for (int idx = tid; idx < NFRAG * 32 * 16; idx += WAVES * 32) {
        int f  = idx >> 9;
        int l  = (idx >> 4) & 31;
        int e  = idx & 15;
        int m  = f >> 1, kc = f & 1;
        int q  = m >> 2, j  = m & 3;
        int Ml = l & 15, lh = l >> 4;
        int v  = e >> 1, p  = e & 1;
        int kl = lh * 16 + v * 2 + p;
        int K  = kc * 32 + kl;
        int row = 16 * j + Ml;      // row within quadrant [0,64)
        float val = 0.f;
        if (row < HID) {
            int gi = HID * q + row; // torch gate order: i,f,g,o
            if (K < HID)           val = W_hh[gi * HID + K];
            else if (K == HID)     val = W_ih[gi];
            else if (K == HID + 1) val = b_ih[gi] + b_hh[gi];
            if (q != 2) val *= 0.5f;   // half-scale i,f,o (sigmoid-via-tanh)
        }
        astage[f][l][e] = (_Float16)val;
    }
    __syncthreads();

    // ---- hoist A to registers (256 VGPRs), resident for entire time loop ----
    v16h A[NFRAG];
    #pragma unroll
    for (int f = 0; f < NFRAG; ++f) {
        union { v16h v; v8h h[2]; } u;
        u.h[0] = *(const v8h*)&astage[f][lane][0];
        u.h[1] = *(const v8h*)&astage[f][lane][8];
        A[f] = u.v;
    }

    const int    batch = (blockIdx.x * WAVES + w) * 16 + n;
    const float* xp    = x + (size_t)batch * T;

    float c[4][8];
    #pragma unroll
    for (int j = 0; j < 4; ++j)
        #pragma unroll
        for (int r = 0; r < 8; ++r) c[j][r] = 0.f;

    // packed f16 h-state in D layout: pj[j][k] = (h[16j+8hl+2k], h[..+1])
    uint32_t pj[4][4];
    #pragma unroll
    for (int j = 0; j < 4; ++j)
        #pragma unroll
        for (int k = 0; k < 4; ++k) pj[j][k] = 0u;

    float xv = xp[0];   // all lanes load (lane n / n+16 duplicate; cache hit)

#ifndef HAVE_PLSWAP
    if (lane < 16) {
        for (int k = 0; k < HROW; ++k) hbuf[w][n][k] = (_Float16)0.f;
        hbuf[w][n][HID + 1] = (_Float16)1.f;
    }
#endif

    for (int t = 0; t < T; ++t) {
        union { v16h v; uint32_t u[8]; } B0u, B1u;

#ifdef HAVE_PLSWAP
        // ---- build B fragments entirely in registers ----
        // B0: K=0..31  <- rows of j=0 / j=1 packs, halves swapped across rows
        // B1: K=32..63 <- j=2 packs and {h48,h49 | x,1 | 0,0} tail
        uint32_t t3[4];
        t3[0] = pj[3][0];          // (h48,h49) on hl=0 rows; zeros on hl=1
        t3[1] = pkh(xv, 1.0f);     // K=50,51 (leak into K=58,59 hits zero A cols)
        t3[2] = 0u;
        t3[3] = 0u;
        #pragma unroll
        for (int k = 0; k < 4; ++k) {
            auto r0 = __builtin_amdgcn_permlane16_swap(pj[0][k], pj[1][k], false, false);
            B0u.u[k]     = r0[0];   // {row0(p0), row0(p1)}
            B0u.u[k + 4] = r0[1];   // {row1(p0), row1(p1)}
            auto r1 = __builtin_amdgcn_permlane16_swap(pj[2][k], t3[k], false, false);
            B1u.u[k]     = r1[0];
            B1u.u[k + 4] = r1[1];
        }
#else
        // ---- fallback: repack via LDS round-trip ----
        hbuf[w][n][HID] = (_Float16)xv;
        #pragma unroll
        for (int j = 0; j < 4; ++j)
            #pragma unroll
            for (int k = 0; k < 4; ++k) {
                int hh0 = 16 * j + 8 * hl + 2 * k;
                if (hh0 < HID) *(uint32_t*)&hbuf[w][n][hh0] = pj[j][k];
            }
        asm volatile("" ::: "memory");
        {
            union { v16h v; v8h h[2]; } ub0, ub1;
            const int kb = hl * 16;
            ub0.h[0] = *(const v8h*)&hbuf[w][n][kb];
            ub0.h[1] = *(const v8h*)&hbuf[w][n][kb + 8];
            ub1.h[0] = *(const v8h*)&hbuf[w][n][32 + kb];
            ub1.h[1] = *(const v8h*)&hbuf[w][n][32 + kb + 8];
            B0u.v = ub0.v; B1u.v = ub1.v;
        }
#endif
        v16h B0 = B0u.v, B1 = B1u.v;

        // prefetch next x (L2-resident; latency hidden behind this step)
        {
            int tn = (t + 1 < T) ? (t + 1) : t;
            xv = xp[tn];
        }

        #pragma unroll
        for (int j = 0; j < 4; ++j) {
            v8f acc[4];
            #pragma unroll
            for (int q = 0; q < 4; ++q) {
                v8f z = {0.f, 0.f, 0.f, 0.f, 0.f, 0.f, 0.f, 0.f};
                z = __builtin_amdgcn_wmma_f32_16x16x32_f16(
                        false, A[(q * 4 + j) * 2],     false, B0, (short)0, z, false, false);
                z = __builtin_amdgcn_wmma_f32_16x16x32_f16(
                        false, A[(q * 4 + j) * 2 + 1], false, B1, (short)0, z, false, false);
                acc[q] = z;
            }
            // elementwise LSTM update; padding slots are EXEC-skipped.
            float hv[8];
            #pragma unroll
            for (int r = 0; r < 8; ++r) hv[r] = 0.f;   // keep pad slots NaN-free
            #pragma unroll
            for (int r = 0; r < 8; ++r) {
                int hh = 16 * j + 8 * hl + r;
                if (hh < HID) {
                    float ig = lstm_sigmoid_half(acc[0][r]);
                    float fg = lstm_sigmoid_half(acc[1][r]);
                    float gg = lstm_tanh(acc[2][r]);
                    float og = lstm_sigmoid_half(acc[3][r]);
                    float cn = fg * c[j][r] + ig * gg;
                    c[j][r]  = cn;
                    hv[r]    = og * lstm_tanh(cn);
                }
            }
            #pragma unroll
            for (int k = 0; k < 4; ++k)
                pj[j][k] = pkh(hv[2 * k], hv[2 * k + 1]);
        }
    }

    // ---- epilogue: dump final h to LDS, then out[b] = h . W_fc + b_fc ----
    #pragma unroll
    for (int j = 0; j < 4; ++j)
        #pragma unroll
        for (int k = 0; k < 4; ++k) {
            int hh0 = 16 * j + 8 * hl + 2 * k;
            if (hh0 < HID) *(uint32_t*)&hbuf[w][n][hh0] = pj[j][k];
        }
    asm volatile("" ::: "memory");
    if (lane < 16) {
        float o = b_fc[0];
        for (int k = 0; k < HID; ++k)
            o += (float)hbuf[w][n][k] * W_fc[k];
        out[batch] = o;
    }
}

extern "C" void kernel_launch(void* const* d_in, const int* in_sizes, int n_in,
                              void* d_out, int out_size, void* d_ws, size_t ws_size,
                              hipStream_t stream) {
    const float* x    = (const float*)d_in[0];
    const float* W_ih = (const float*)d_in[1];
    const float* W_hh = (const float*)d_in[2];
    const float* b_ih = (const float*)d_in[3];
    const float* b_hh = (const float*)d_in[4];
    const float* W_fc = (const float*)d_in[5];
    const float* b_fc = (const float*)d_in[6];
    float* out = (float*)d_out;

    const int B = out_size;                 // 8192 (OUT=1)
    const int T = in_sizes[0] / B;          // 2048 (INPUT=1)
    const int grid = B / (WAVES * 16);      // 16 batch rows per wave

    lstm_wmma_kernel<<<grid, WAVES * 32, 0, stream>>>(
        x, W_ih, W_hh, b_ih, b_hh, W_fc, b_fc, out, T);
}
